// MultiHeadAttention_58969900974210
// MI455X (gfx1250) — matrix operand order repaired and checked by reading, not compiled, not run on hardware
//
#include <hip/hip_runtime.h>
#include <hip/hip_bf16.h>

// ---------------------------------------------------------------------------
// MHA for MI455X (gfx1250, wave32).  ~137 GFLOP vs ~110MB traffic -> hard
// compute-bound: all GEMMs on v_wmma_f32_16x16x32_bf16 (f32 accum).
// Flash attention with Tensor-Data-Mover double-buffered K/V staging in LDS
// (s_wait_tensorcnt sync); softmax row-sums via WMMA against all-ones B.
// ---------------------------------------------------------------------------

typedef __bf16 bf16_t;
typedef __attribute__((ext_vector_type(16))) __bf16        v16bf;
typedef __attribute__((ext_vector_type(8)))  __bf16        v8bf;
typedef __attribute__((ext_vector_type(8)))  float         v8f;
typedef __attribute__((ext_vector_type(4)))  unsigned int  u32x4;
typedef __attribute__((ext_vector_type(8)))  int           i32x8;
typedef __attribute__((ext_vector_type(4)))  int           i32x4;

#define B_SZ  4
#define C_SZ  1024
#define T_SZ  2048
#define H_SZ  16
#define KD_SZ 64

__device__ __forceinline__ v16bf cat8(v8bf a, v8bf b) {
  return __builtin_shufflevector(a, b, 0,1,2,3,4,5,6,7,8,9,10,11,12,13,14,15);
}

// A-matrix 16x32 bf16 fragment.  Lane holds row M=(lane&15); element e ->
// K = e + 8*(e>=8) + 8*(lane>=16): two 8-elem runs at +8*hi and +16+8*hi.
__device__ __forceinline__ v16bf load_a_frag(const bf16_t* rowbase, int lane) {
  const int off = ((lane >> 4) & 1) * 8;
  v8bf lo = *(const v8bf*)(rowbase + off);
  v8bf hi = *(const v8bf*)(rowbase + off + 16);
  return cat8(lo, hi);
}

// B-matrix 32x16 bf16 fragment.  Lane holds column N=(lane&15); element e ->
// K = e + 8*(e>=8) + 8*(lane>=16): two 8-elem runs at +8*hi and +16+8*hi.
__device__ __forceinline__ v16bf load_b_frag(const bf16_t* colbase, int lane) {
  const int off = ((lane >> 4) & 1) * 8;
  v8bf lo = *(const v8bf*)(colbase + off);
  v8bf hi = *(const v8bf*)(colbase + off + 16);
  return cat8(lo, hi);
}

// ---------------------------------------------------------------------------
// TDM: DMA one 64-row x 128B tile (global, row stride row_stride_8B x 8B)
// into LDS at lds_dst.  data_size=8B; LDS pad 16B every 128B -> 144B rows
// (stride 72 bf16): 36-bank stride, conflict-free for 16-lane frag reads.
// D# group0: count=1 | lds_addr | global_addr | type=2.
// D# group1: data_size=3, pad_enable, pad_interval=4 (128B), pad_amount=3 (16B),
//            tensor_dim0=512, tensor_dim1=64, tile_dim0=16, tile_dim1=64, stride0.
// Toolchain uses the 6-arg builtin: (g0, g1, g2, g3, g4, cpol).
// ---------------------------------------------------------------------------
__device__ __forceinline__ void tdm_load_tile(const bf16_t* gsrc, bf16_t* lds_dst,
                                              unsigned row_stride_8B) {
  const unsigned long long ga = (unsigned long long)(size_t)gsrc;
  u32x4 g0;
  g0[0] = 1u;                                        // count=1 (valid user D#)
  g0[1] = (unsigned)(size_t)lds_dst;                 // LDS byte address (addr[31:0])
  g0[2] = (unsigned)(ga & 0xFFFFFFFFu);              // global_addr[31:0]
  g0[3] = (unsigned)((ga >> 32) & 0x1FFFFFFu) | (2u << 30);  // global_addr[56:32] | type=2
  i32x8 g1;
  g1[0] = (3 << 16) | (1 << 20) | (4 << 22) | (3 << 25);  // 8B elems, pad 16B/128B
  g1[1] = (int)(512u << 16);                         // tensor_dim0 = 512 (8B units)
  g1[2] = (int)(64u << 16);                          // tensor_dim1 = 64
  g1[3] = (int)(16u << 16);                          // tile_dim0 = 16 (= 128B row)
  g1[4] = 64;                                        // tile_dim1 = 64 rows
  g1[5] = (int)row_stride_8B;                        // tensor_dim0_stride (8B units)
  g1[6] = 0;
  g1[7] = 0;
  i32x4 z4 = {0, 0, 0, 0};
  i32x8 z8 = {0, 0, 0, 0, 0, 0, 0, 0};
  __builtin_amdgcn_tensor_load_to_lds(g0, g1, z4, z4, z8, 0);
}

// ---------------------------------------------------------------------------
// 1x1-conv GEMM: Y[b,o,t] = (sum_c W[o,c]*X[b,c,t] + bias[o]) * oscale
// Block = 256 thr (8 waves), tile 128(M) x 256(N), K-step 32.
// Wave grid 2(M) x 4(N): each wave owns 64x64 = 4x4 WMMA fragments.
// MODE 0: bf16 [B,H,T,64] (q/k staging) -- rows of a D-frag are 8 consecutive
//         kd at fixed t -> one packed 16B store per fragment.
// MODE 2: bf16 [B,C,T] (v staging);  MODE 3: f32 [B,C,T] (final output)
// ---------------------------------------------------------------------------
template <typename TX, int MODE>
__global__ __launch_bounds__(256) void gemm1x1_kernel(
    const float* __restrict__ W,     // [1024 x 1024] row-major (o, c)
    const TX*    __restrict__ X,     // [B][1024][T]
    const float* __restrict__ bias,  // [1024]
    void*        __restrict__ out,
    float oscale)
{
  constexpr int BM = 128, BN = 256, BK = 32, LDSTR = 40;  // 80B rows: 16B aligned, conflict-free
  __shared__ __align__(16) bf16_t As[BM * LDSTR];  // [m][k]
  __shared__ __align__(16) bf16_t Bs[BN * LDSTR];  // [n][k]  (transposed on fill)

  const int tid  = threadIdx.x;
  const int lane = tid & 31;
  const int wave = tid >> 5;
  const int wm   = wave & 1;    // 0..1 -> 64 rows each
  const int wn   = wave >> 1;   // 0..3 -> 64 cols each
  const int b    = blockIdx.z;
  const int m0   = blockIdx.x * BM;
  const int n0   = blockIdx.y * BN;

  const TX* Xp = X + (size_t)b * C_SZ * T_SZ;

  v8f acc[4][4] = {};

  for (int kc = 0; kc < C_SZ; kc += BK) {
    {  // stage A: W tile [128m x 32k] fp32 -> bf16 LDS [m][k]
      const int row = tid >> 1, cs = (tid & 1) * 16;
      const float* src = W + (size_t)(m0 + row) * C_SZ + kc + cs;
      bf16_t* dst = &As[row * LDSTR + cs];
#pragma unroll
      for (int j = 0; j < 16; ++j) dst[j] = (bf16_t)src[j];
    }
    {  // stage B: X tile [32k x 256n] -> transposed bf16 LDS [n][k]
      const int crow = tid >> 3, ts = (tid & 7) * 32;
      const TX* src = Xp + (size_t)(kc + crow) * T_SZ + n0 + ts;
#pragma unroll
      for (int j = 0; j < 32; ++j)
        Bs[(ts + j) * LDSTR + crow] = (bf16_t)(float)src[j];
    }
    __syncthreads();

    v16bf bfrag[4];
#pragma unroll
    for (int nf = 0; nf < 4; ++nf) {
      const int n = wn * 64 + nf * 16 + (lane & 15);
      bfrag[nf] = load_b_frag(&Bs[n * LDSTR], lane);
    }
#pragma unroll
    for (int mf = 0; mf < 4; ++mf) {
      const int m = wm * 64 + mf * 16 + (lane & 15);
      v16bf afrag = load_a_frag(&As[m * LDSTR], lane);
#pragma unroll
      for (int nf = 0; nf < 4; ++nf)
        acc[mf][nf] = __builtin_amdgcn_wmma_f32_16x16x32_bf16(
            false, afrag, false, bfrag[nf], (short)0, acc[mf][nf], false, false);
    }
    __syncthreads();
  }

  // ---- epilogue: D layout -> lane holds N=(lane&15), VGPR r -> M=r+8*(lane>=16)
  const int hi = (lane >> 4) & 1;
#pragma unroll
  for (int mf = 0; mf < 4; ++mf) {
#pragma unroll
    for (int nf = 0; nf < 4; ++nf) {
      const int t = n0 + wn * 64 + nf * 16 + (lane & 15);
      const int obase = m0 + wm * 64 + mf * 16 + 8 * hi;  // o = obase + r
      if (MODE == 0) {  // packed: 8 consecutive kd at fixed t
        const int h = obase >> 6, kd0 = obase & 63;  // kd0 multiple of 8 -> 16B aligned
        v8bf pk;
#pragma unroll
        for (int r = 0; r < 8; ++r)
          pk[r] = (bf16_t)((acc[mf][nf][r] + bias[obase + r]) * oscale);
        *(v8bf*)&((bf16_t*)out)[(((size_t)b * H_SZ + h) * T_SZ + t) * KD_SZ + kd0] = pk;
      } else {
#pragma unroll
        for (int r = 0; r < 8; ++r) {
          const int o = obase + r;
          const float val = (acc[mf][nf][r] + bias[o]) * oscale;
          if (MODE == 3) ((float*)out)[((size_t)b * C_SZ + o) * T_SZ + t] = val;
          else           ((bf16_t*)out)[((size_t)b * C_SZ + o) * T_SZ + t] = (bf16_t)val;
        }
      }
    }
  }
}

// ---------------------------------------------------------------------------
// Flash attention.  Grid: (B*H, T/128).  8 waves; wave owns 16 query rows.
// All waves share (b,h) and the S-stream, so K/V S-blocks (64 x 64 bf16) are
// staged once per block into LDS by the Tensor Data Mover, double-buffered:
// wave 0 issues next-block DMAs, computes overlap, s_wait_tensorcnt + barriers
// sequence the buffers.  Per S-block: 8 score + 8 PV + 2 row-sum WMMAs/wave.
// ---------------------------------------------------------------------------
__global__ __launch_bounds__(256) void attn_kernel(
    const bf16_t* __restrict__ qT,  // [B*H][T][64]  (pre-scaled by 1/8*log2e)
    const bf16_t* __restrict__ kT,  // [B*H][T][64]
    const bf16_t* __restrict__ vm,  // [B*H][64][T]
    bf16_t*       __restrict__ ao)  // [B][1024][T]
{
  constexpr int VSTR = 72;  // padded LDS row stride (bf16): 144B, conflict-free
  __shared__ __align__(16) bf16_t KL[2][64 * VSTR];   // [s_local][c]
  __shared__ __align__(16) bf16_t VL[2][64 * VSTR];   // [kd][s_local]
  __shared__ __align__(16) bf16_t Pst[8][16][64];     // per-wave P staging

  const int tid  = threadIdx.x;
  const int lane = tid & 31;
  const int wave = tid >> 5;
  const int hi   = (lane >> 4) & 1;
  const int bh   = blockIdx.x;
  const int t0   = blockIdx.y * 128 + wave * 16;

  const bf16_t* kplane = kT + (size_t)bh * T_SZ * KD_SZ;  // [s][64]
  const bf16_t* vplane = vm + (size_t)bh * KD_SZ * T_SZ;  // [kd][T]

  // ---- prologue: DMA S-block 0 into buffer 0
  if (wave == 0) {
    tdm_load_tile(kplane, &KL[0][0], 16);                    // contiguous 128B rows
    tdm_load_tile(vplane, &VL[0][0], (unsigned)(T_SZ / 4));  // row stride T*2B = 512 x 8B
  }

  // Q fragments for this wave's 16 rows (head-dim chunks 0..31, 32..63)
  const bf16_t* qrow = qT + ((size_t)bh * T_SZ + t0 + (lane & 15)) * KD_SZ;
  const v16bf aq0 = load_a_frag(qrow, lane);
  const v16bf aq1 = load_a_frag(qrow + 32, lane);

  v16bf onesf;
#pragma unroll
  for (int i = 0; i < 16; ++i) onesf[i] = (bf16_t)1.0f;

  v8f oacc[4] = {};
  v8f lsum = {};
  float mrun[8];
#pragma unroll
  for (int r = 0; r < 8; ++r) mrun[r] = -1e30f;

  for (int i = 0; i < T_SZ / 64; ++i) {
    const int buf = i & 1;
    if (wave == 0) {
      if (i + 1 < T_SZ / 64) {  // DMA next S-block into the other buffer
        const int sn = (i + 1) * 64;
        tdm_load_tile(kplane + (size_t)sn * KD_SZ, &KL[buf ^ 1][0], 16);
        tdm_load_tile(vplane + sn, &VL[buf ^ 1][0], (unsigned)(T_SZ / 4));
        __builtin_amdgcn_s_wait_tensorcnt(2);  // current block's pair landed
      } else {
        __builtin_amdgcn_s_wait_tensorcnt(0);
      }
    }
    __syncthreads();  // current buffer visible to all waves

    const bf16_t* Kb = &KL[buf][0];
    const bf16_t* Vb = &VL[buf][0];

    // ---- scores: four 16-wide s-tiles, contraction over head-dim 64 (2 chunks)
    v8f sc[4] = {};
#pragma unroll
    for (int sb = 0; sb < 4; ++sb) {
      const bf16_t* krow = Kb + (sb * 16 + (lane & 15)) * VSTR;
      const v16bf bk0 = load_b_frag(krow, lane);
      const v16bf bk1 = load_b_frag(krow + 32, lane);
      sc[sb] = __builtin_amdgcn_wmma_f32_16x16x32_bf16(
          false, aq0, false, bk0, (short)0, sc[sb], false, false);
      sc[sb] = __builtin_amdgcn_wmma_f32_16x16x32_bf16(
          false, aq1, false, bk1, (short)0, sc[sb], false, false);
    }

    // ---- online softmax (rows M=r+8*hi live across lanes of each 16-lane half)
#pragma unroll
    for (int r = 0; r < 8; ++r) {
      float m = fmaxf(fmaxf(sc[0][r], sc[1][r]), fmaxf(sc[2][r], sc[3][r]));
#pragma unroll
      for (int off = 1; off < 16; off <<= 1) m = fmaxf(m, __shfl_xor(m, off, 32));
      const float mnew = fmaxf(mrun[r], m);
      const float corr = __builtin_amdgcn_exp2f(mrun[r] - mnew);
      mrun[r] = mnew;
#pragma unroll
      for (int sb = 0; sb < 4; ++sb) {
        const float p = __builtin_amdgcn_exp2f(sc[sb][r] - mnew);
        Pst[wave][r + 8 * hi][sb * 16 + (lane & 15)] = (bf16_t)p;
      }
      lsum[r] *= corr;
#pragma unroll
      for (int nf = 0; nf < 4; ++nf) oacc[nf][r] *= corr;
    }

    __asm__ volatile("s_wait_dscnt 0x0" ::: "memory");  // wave-private LDS RAW

    // ---- PV + row-sum: A = P (16t x 64s, two frags) from LDS
    const v16bf ap0 = load_a_frag(&Pst[wave][lane & 15][0], lane);
    const v16bf ap1 = load_a_frag(&Pst[wave][lane & 15][32], lane);
    lsum = __builtin_amdgcn_wmma_f32_16x16x32_bf16(
        false, ap0, false, onesf, (short)0, lsum, false, false);
    lsum = __builtin_amdgcn_wmma_f32_16x16x32_bf16(
        false, ap1, false, onesf, (short)0, lsum, false, false);
#pragma unroll
    for (int nf = 0; nf < 4; ++nf) {
      const bf16_t* vrow = Vb + (nf * 16 + (lane & 15)) * VSTR;
      const v16bf bv0 = load_b_frag(vrow, lane);
      const v16bf bv1 = load_b_frag(vrow + 32, lane);
      oacc[nf] = __builtin_amdgcn_wmma_f32_16x16x32_bf16(
          false, ap0, false, bv0, (short)0, oacc[nf], false, false);
      oacc[nf] = __builtin_amdgcn_wmma_f32_16x16x32_bf16(
          false, ap1, false, bv1, (short)0, oacc[nf], false, false);
    }
    __syncthreads();  // all reads done before next iteration's DMA overwrites
  }

  // ---- normalize + store: rows r are 8 consecutive t at fixed kd -> 16B store
#pragma unroll
  for (int nf = 0; nf < 4; ++nf) {
    const int kd = nf * 16 + (lane & 15);
    v8bf pk;
#pragma unroll
    for (int r = 0; r < 8; ++r) pk[r] = (bf16_t)(oacc[nf][r] / lsum[r]);
    *(v8bf*)&ao[((size_t)bh * KD_SZ + kd) * T_SZ + t0 + 8 * hi] = pk;
  }
}

// ---------------------------------------------------------------------------
extern "C" void kernel_launch(void* const* d_in, const int* in_sizes, int n_in,
                              void* d_out, int out_size, void* d_ws, size_t ws_size,
                              hipStream_t stream) {
  const float* x  = (const float*)d_in[0];
  const float* c  = (const float*)d_in[1];
  const float* Wq = (const float*)d_in[2];
  const float* bq = (const float*)d_in[3];
  const float* Wk = (const float*)d_in[4];
  const float* bk = (const float*)d_in[5];
  const float* Wv = (const float*)d_in[6];
  const float* bv = (const float*)d_in[7];
  const float* Wo = (const float*)d_in[8];
  const float* bo = (const float*)d_in[9];
  float* out = (float*)d_out;

  const size_t NELT = (size_t)B_SZ * C_SZ * T_SZ;  // 8,388,608 (= B*H*T*64)
  bf16_t* qT = (bf16_t*)d_ws;         // [B,H,T,64]
  bf16_t* kS = qT + NELT;             // [B,H,T,64]
  bf16_t* vS = kS + NELT;             // [B,H,64,T]
  bf16_t* ao = vS + NELT;             // [B,C,T]

  const float QSCL = 0.125f * 1.44269504089f;  // 1/sqrt(64) * log2(e), folded into q

  dim3 blk(256);
  dim3 g_gemm(C_SZ / 128, T_SZ / 256, B_SZ);
  dim3 g_attn(B_SZ * H_SZ, T_SZ / 128);

  gemm1x1_kernel<float, 0><<<g_gemm, blk, 0, stream>>>(Wq, x, bq, (void*)qT, QSCL);
  gemm1x1_kernel<float, 0><<<g_gemm, blk, 0, stream>>>(Wk, c, bk, (void*)kS, 1.0f);
  gemm1x1_kernel<float, 2><<<g_gemm, blk, 0, stream>>>(Wv, c, bv, (void*)vS, 1.0f);
  attn_kernel<<<g_attn, blk, 0, stream>>>(qT, kS, vS, ao);
  gemm1x1_kernel<bf16_t, 3><<<g_gemm, blk, 0, stream>>>(Wo, ao, bo, (void*)out, 1.0f);
}
